// GAT_36653250904099
// MI455X (gfx1250) — matrix operand order repaired and checked by reading, not compiled, not run on hardware
//
#include <hip/hip_runtime.h>
#include <hip/hip_fp16.h>

typedef __attribute__((ext_vector_type(16))) _Float16 v16h;
typedef __attribute__((ext_vector_type(4)))  _Float16 v4h;
typedef __attribute__((ext_vector_type(8)))  float    v8f;

#define N_NODES 4096
#define T_DIM   4
#define D_DIM   128
#define SCALE   16.0f   // sqrt(2*128)

// ---------------------------------------------------------------------------
// Kernel 0: pack W (f32 row-major [K][Ncol]) into f16 WMMA B-operand layout.
// B layout (16x16x32 f16): lane L holds column N = (L&15); K = 8*(L>>4) + j + (j&8)
// for halves j=0..15 (packed 2 per dword, ascending).
// WB index: [kc(4)][dt(8)][lane(32)][j(16)]
// ---------------------------------------------------------------------------
__global__ void __launch_bounds__(256) pack_W_kernel(const float* __restrict__ W,
                                                     _Float16* __restrict__ WB) {
  int tid  = blockIdx.x * 256 + threadIdx.x;   // 0..1023
  int lane = tid & 31;
  int dt   = (tid >> 5) & 7;
  int kc   = tid >> 8;
  int n    = dt * 16 + (lane & 15);
  int kb   = (lane & 16) ? 8 : 0;
  v16h v;
#pragma unroll
  for (int j = 0; j < 16; ++j) {
    int K = kc * 32 + kb + j + (j & 8);
    v[j] = (_Float16)W[(size_t)K * D_DIM + n];
  }
  *(v16h*)(WB + (size_t)tid * 16) = v;
}

// ---------------------------------------------------------------------------
// Kernel 1: Wx = x @ W via v_wmma_f32_16x16x32_f16.
// One wave computes a 16(row)x128(col) strip for a fixed t.
// A layout (16x32 f16): lane L holds row M=(L&15); halves j=0..7 -> K=akb+j,
// j=8..15 -> K=akb+16+(j-8), akb = (L>=16)?8:0.
// Outputs: WxR row-major f16 [t][m][d], and WxB in B-operand packed layout
// [t][mc(128)][dt(8)][lane(32)][j(16)] for stage 2.
// ---------------------------------------------------------------------------
__global__ void __launch_bounds__(128) gemm_xW_kernel(const float* __restrict__ x,
                                                      const _Float16* __restrict__ WB,
                                                      _Float16* __restrict__ WxR,
                                                      _Float16* __restrict__ WxB) {
  int lane = threadIdx.x & 31;
  int wave = threadIdx.x >> 5;
  int tile = blockIdx.x * 4 + wave;     // 0..1023
  int t    = tile >> 8;                 // 256 row-tiles per t
  int n0   = (tile & 255) * 16;
  int hi   = lane >> 4;
  int akb  = hi ? 8 : 0;

  v8f acc[8];
#pragma unroll
  for (int dt = 0; dt < 8; ++dt) acc[dt] = (v8f){0,0,0,0,0,0,0,0};

  const float* xrow = x + ((size_t)(n0 + (lane & 15)) * T_DIM + t) * D_DIM;
  const v16h*  WBv  = (const v16h*)WB;

#pragma unroll
  for (int kc = 0; kc < 4; ++kc) {
    const float* p = xrow + kc * 32 + akb;
    float4 r0 = *(const float4*)(p);
    float4 r1 = *(const float4*)(p + 4);
    float4 r2 = *(const float4*)(p + 16);
    float4 r3 = *(const float4*)(p + 20);
    v16h a;
    a[0]=(_Float16)r0.x;  a[1]=(_Float16)r0.y;  a[2]=(_Float16)r0.z;  a[3]=(_Float16)r0.w;
    a[4]=(_Float16)r1.x;  a[5]=(_Float16)r1.y;  a[6]=(_Float16)r1.z;  a[7]=(_Float16)r1.w;
    a[8]=(_Float16)r2.x;  a[9]=(_Float16)r2.y;  a[10]=(_Float16)r2.z; a[11]=(_Float16)r2.w;
    a[12]=(_Float16)r3.x; a[13]=(_Float16)r3.y; a[14]=(_Float16)r3.z; a[15]=(_Float16)r3.w;
#pragma unroll
    for (int dt = 0; dt < 8; ++dt) {
      v16h b = WBv[(kc * 8 + dt) * 32 + lane];
      acc[dt] = __builtin_amdgcn_wmma_f32_16x16x32_f16(
          false, a, false, b, (short)0, acc[dt], false, false);
    }
  }

  // C/D layout: VGPR i holds row M = i + 8*hi, col N = dt*16 + (lane&15).
  int mc  = n0 >> 5;          // 32-row chunk index for WxB
  int klb = (n0 & 16);        // this tile covers K-local klb..klb+15
#pragma unroll
  for (int dt = 0; dt < 8; ++dt) {
#pragma unroll
    for (int i = 0; i < 8; ++i) {
      int m = n0 + i + 8 * hi;
      _Float16 h = (_Float16)acc[dt][i];
      // row-major copy
      WxR[((size_t)t * N_NODES + m) * D_DIM + dt * 16 + (lane & 15)] = h;
      // B-operand packed copy
      int kl    = klb + i + 8 * hi;           // K-local in [0,32)
      int lanep = (lane & 15) + (kl & 16);    // lanes 16-31 hold K 16..31
      int jp    = kl & 15;
      WxB[((((size_t)t * 128 + mc) * 8 + dt) * 32 + lanep) * 16 + jp] = h;
    }
  }
}

// ---------------------------------------------------------------------------
// Kernel 2: s1[t][m] = (Wx[t][m][:] . a1)/16 ; s2 likewise. One wave per row.
// ---------------------------------------------------------------------------
__global__ void __launch_bounds__(256) svec_kernel(const _Float16* __restrict__ WxR,
                                                   const float* __restrict__ a1,
                                                   const float* __restrict__ a2,
                                                   float* __restrict__ s1,
                                                   float* __restrict__ s2) {
  int lane = threadIdx.x & 31;
  int r    = blockIdx.x * 8 + (threadIdx.x >> 5);   // 0..16383 = t*N + m
  v4h   h  = *(const v4h*)(WxR + (size_t)r * D_DIM + lane * 4);
  float4 w1 = *(const float4*)(a1 + lane * 4);
  float4 w2 = *(const float4*)(a2 + lane * 4);
  float f0 = (float)h[0], f1 = (float)h[1], f2 = (float)h[2], f3 = (float)h[3];
  float d1 = f0 * w1.x + f1 * w1.y + f2 * w1.z + f3 * w1.w;
  float d2 = f0 * w2.x + f1 * w2.y + f2 * w2.z + f3 * w2.w;
#pragma unroll
  for (int off = 16; off > 0; off >>= 1) {
    d1 += __shfl_xor(d1, off);
    d2 += __shfl_xor(d2, off);
  }
  if (lane == 0) {
    s1[r] = d1 * (1.0f / SCALE);
    s2[r] = d2 * (1.0f / SCALE);
  }
}

// ---------------------------------------------------------------------------
// Kernel 3: flash-style attention + aggregation + residual.
// Block = 256 thr (8 waves); each wave owns a 16-row n-tile for one t.
// The 8KB packed-B chunk per m-step is shared by all 8 waves: it is staged
// into LDS with GLOBAL_LOAD_ASYNC_TO_LDS_B128 (ASYNCcnt), 4 buffers,
// prefetch distance 2, one workgroup barrier per iteration.
//   - chunk mc+2's target buffer was last read at iter mc-2; every wave's
//     arrival at barrier mc-1 orders those reads before any issue at iter mc.
// Row max of score is lrelu(max_m s1 + s2[n]) since lrelu is monotone, so a
// single pass accumulates Z (VALU, co-executes with XDL) and P*Wx (WMMA).
// ---------------------------------------------------------------------------
__global__ void __launch_bounds__(256) attn_kernel(const float* __restrict__ x,
                                                   const _Float16* __restrict__ WxB,
                                                   const float* __restrict__ s1g,
                                                   const float* __restrict__ s2g,
                                                   float* __restrict__ out) {
  __shared__ float ls1[N_NODES];
  __shared__ float red[256];
  __shared__ __align__(32) char bufs[4 * 8192];   // 4 x 8KB B-chunk buffers

  int tid  = threadIdx.x;
  int lane = tid & 31;
  int wave = tid >> 5;
  int t    = blockIdx.x >> 5;                 // 32 blocks per t
  int n0   = (blockIdx.x & 31) * 128 + wave * 16;

  // stage s1[t][:] into LDS and find its max
  float lmax = -1e30f;
  for (int i = tid; i < N_NODES; i += 256) {
    float v = s1g[t * N_NODES + i];
    ls1[i] = v;
    lmax = fmaxf(lmax, v);
  }
  red[tid] = lmax;
  __syncthreads();
  for (int s = 128; s > 0; s >>= 1) {
    if (tid < s) red[tid] = fmaxf(red[tid], red[tid + s]);
    __syncthreads();
  }
  float max1 = red[0];

  int   hi  = lane >> 4;
  int   akb = hi ? 8 : 0;
  float s2n = s2g[t * N_NODES + n0 + (lane & 15)];
  float mn0 = max1 + s2n;
  float Mn  = fmaxf(mn0, 0.01f * mn0);        // lrelu(max1 + s2n) = row max

  v8f acc[8];
#pragma unroll
  for (int dt = 0; dt < 8; ++dt) acc[dt] = (v8f){0,0,0,0,0,0,0,0};
  float zsum = 0.0f;

  // Async staging: each thread owns a 32B slice of the 8KB chunk (2 x b128).
  const char* gbase   = (const char*)WxB + (size_t)t * 128 * 8192;
  uint32_t    ldsbase = (uint32_t)(uintptr_t)(&bufs[0]);  // LDS byte address

  auto issue_chunk = [&](int c) {
    const char* src = gbase + (size_t)c * 8192 + tid * 32;
    uint32_t    dst = ldsbase + (uint32_t)(((c & 3) * 8192) + tid * 32);
    asm volatile("global_load_async_to_lds_b128 %0, %1, off"
                 :: "v"(dst), "v"(src) : "memory");
    asm volatile("global_load_async_to_lds_b128 %0, %1, off offset:16"
                 :: "v"(dst), "v"(src) : "memory");
  };

  issue_chunk(0);
  issue_chunk(1);

  for (int mc = 0; mc < 128; ++mc) {
    if (mc + 2 < 128) issue_chunk(mc + 2);
    if (mc < 126) {
      asm volatile("s_wait_asynccnt 0x4" ::: "memory");  // chunk mc retired
    } else {
      asm volatile("s_wait_asynccnt 0x0" ::: "memory");
    }
    __syncthreads();   // all waves' slices of chunk mc are in LDS

    v16h a;
    float zc = 0.0f;
#pragma unroll
    for (int j = 0; j < 16; ++j) {
      // A layout: K = 16*hi + j  (contiguous, matches WxB packing)
      float s1v = ls1[mc * 32 + 16 * hi + j];
      float sc  = s1v + s2n;
      sc = fmaxf(sc, 0.01f * sc);             // leaky_relu, slope 0.01
      float p = __expf(sc - Mn);
      zc += p;
      a[j] = (_Float16)p;
    }
    zsum += zc;

    const v16h* Bp = (const v16h*)(bufs + (mc & 3) * 8192);
#pragma unroll
    for (int dt = 0; dt < 8; ++dt) {
      v16h b = Bp[dt * 32 + lane];
      acc[dt] = __builtin_amdgcn_wmma_f32_16x16x32_f16(
          false, a, false, b, (short)0, acc[dt], false, false);
    }
  }

  // Z per row: lanes L and L^16 cover complementary K halves of row (L&15)
  float zrow = zsum + __shfl_xor(zsum, 16);
  float invz[8];
#pragma unroll
  for (int i = 0; i < 8; ++i)
    invz[i] = 1.0f / __shfl(zrow, i + (hi << 3));   // Z for row i + 8*hi

#pragma unroll
  for (int dt = 0; dt < 8; ++dt) {
#pragma unroll
    for (int i = 0; i < 8; ++i) {
      int n = n0 + i + 8 * hi;
      size_t idx = ((size_t)n * T_DIM + t) * D_DIM + dt * 16 + (lane & 15);
      float agg = acc[dt][i] * invz[i];
      agg = fmaxf(agg, 0.01f * agg);                // lrelu on aggregate
      out[idx] = x[idx] - agg;
    }
  }
}

// ---------------------------------------------------------------------------
// Workspace layout (bytes):
//   [0,        32768)               WB   : packed W, f16
//   [32768,    32768+4194304)       WxB  : packed Wx, f16 (B-operand layout)
//   [.. +4194304)                   WxR  : row-major Wx, f16
//   [.. +65536)                     s1   : f32 [T][N]
//   [.. +65536)                     s2   : f32 [T][N]
// Total ~8.2 MB.
// ---------------------------------------------------------------------------
extern "C" void kernel_launch(void* const* d_in, const int* in_sizes, int n_in,
                              void* d_out, int out_size, void* d_ws, size_t ws_size,
                              hipStream_t stream) {
  const float* x  = (const float*)d_in[0];
  const float* W  = (const float*)d_in[1];
  const float* a1 = (const float*)d_in[2];
  const float* a2 = (const float*)d_in[3];
  float* out = (float*)d_out;

  char* ws = (char*)d_ws;
  _Float16* WB  = (_Float16*)(ws);
  _Float16* WxB = (_Float16*)(ws + 32768);
  _Float16* WxR = (_Float16*)(ws + 32768 + 4194304);
  float*    s1  = (float*)(ws + 32768 + 2 * 4194304);
  float*    s2  = (float*)(ws + 32768 + 2 * 4194304 + 65536);

  pack_W_kernel<<<dim3(4),    dim3(256), 0, stream>>>(W, WB);
  gemm_xW_kernel<<<dim3(256), dim3(128), 0, stream>>>(x, WB, WxR, WxB);
  svec_kernel<<<dim3(2048),   dim3(256), 0, stream>>>(WxR, a1, a2, s1, s2);
  attn_kernel<<<dim3(128),    dim3(256), 0, stream>>>(x, WxB, s1, s2, out);
}
